// CustomMultiheadAttention_5042291605574
// MI455X (gfx1250) — matrix operand order repaired and checked by reading, not compiled, not run on hardware
//
#include <hip/hip_runtime.h>
#include <hip/hip_bf16.h>

// ---------------------------------------------------------------------------
// Types for CDNA5 WMMA (wave32, v_wmma_f32_16x16x32_bf16)
// ---------------------------------------------------------------------------
typedef __attribute__((ext_vector_type(16))) __bf16 v16bf;
typedef __attribute__((ext_vector_type(8)))  float  v8f;
typedef __attribute__((ext_vector_type(4)))  unsigned int uint4v;

union Frag {
    v16bf  bf;
    uint4v u[2];
};

__device__ __forceinline__ unsigned short f2bf(float f) {
    union { float f; unsigned int u; } c; c.f = f;
    unsigned int u = c.u;
    unsigned int r = u + 0x7FFFu + ((u >> 16) & 1u); // round-to-nearest-even
    return (unsigned short)(r >> 16);
}
__device__ __forceinline__ unsigned int pack2bf(float a, float b) {
    return (unsigned int)f2bf(a) | ((unsigned int)f2bf(b) << 16);
}

__device__ __forceinline__ v8f wmma_bf16(const Frag& a, const Frag& b, v8f c) {
    return __builtin_amdgcn_wmma_f32_16x16x32_bf16(
        /*neg_a=*/false, a.bf, /*neg_b=*/false, b.bf,
        /*c_mod=*/(short)0, c, /*reuse_a=*/false, /*reuse_b=*/false);
}

// Async global->LDS copy of 32 contiguous bytes (2 x b128), CDNA5 TDM-lite
// path tracked by ASYNCcnt. INST_OFFSET is added to both the global and the
// LDS address, so one address pair serves both chunks.
__device__ __forceinline__ void async_ld32B(unsigned lds_off, const void* gaddr) {
    asm volatile("global_load_async_to_lds_b128 %0, %1, off\n\t"
                 "global_load_async_to_lds_b128 %0, %1, off offset:16"
                 :: "v"(lds_off), "v"(gaddr) : "memory");
}
__device__ __forceinline__ void wait_async0() {
    asm volatile("s_wait_asynccnt 0x0" ::: "memory");
}
__device__ __forceinline__ unsigned lds_offset_of(const void* p) {
    // low 32 bits of a flat LDS address == offset within the group allocation
    return (unsigned)(size_t)p;
}

// ---------------------------------------------------------------------------
// Problem constants
// ---------------------------------------------------------------------------
#define BB 8
#define SS 1024
#define EE 1024
#define HH 16
#define DD 64
#define MM (BB * SS)      // 8192

// workspace offsets (bytes)
#define XB_OFF   0u
#define WQKV_OFF 16777216u
#define WOUT_OFF 23068672u
#define QB_OFF   25165824u
#define KB_OFF   41943040u
#define VT_OFF   58720256u
#define AO_OFF   75497472u

// ---------------------------------------------------------------------------
// f32 -> bf16 conversion (vectorized x4)
// ---------------------------------------------------------------------------
__global__ __launch_bounds__(256) void cvt_f32_bf16(const float* __restrict__ in,
                                                    unsigned short* __restrict__ out,
                                                    int n4) {
    int i = blockIdx.x * blockDim.x + threadIdx.x;
    if (i >= n4) return;
    float4 v = ((const float4*)in)[i];
    uint2 p;
    p.x = pack2bf(v.x, v.y);
    p.y = pack2bf(v.z, v.w);
    ((uint2*)out)[i] = p;
}

// ---------------------------------------------------------------------------
// NT GEMM:  C[M,N] = A[M,K](bf16) * B[N,K](bf16)^T + bias[N]
// mode 0: write f32 C
// mode 1: QKV scatter -> Qb/Kb [B,H,S,D] bf16 ; Vt [B,H,D,S] bf16
// Block = 256 threads (8 waves), tile 128x128, K-step 32.
// Double-buffered LDS filled by global_load_async_to_lds_b128; one barrier
// per k-step (wait own ASYNCcnt -> barrier -> issue next tile -> compute).
// ---------------------------------------------------------------------------
#define LDT 40                 // padded LDS row stride (elems); 80B = 5 x 16B
#define TILE_ELEMS (128 * LDT) // one buffer
#define TILE_BYTES (TILE_ELEMS * 2)

__global__ __launch_bounds__(256) void gemm_bf16_nt(
    const unsigned short* __restrict__ A,
    const unsigned short* __restrict__ Bm,
    const float* __restrict__ bias,
    float* __restrict__ Cout,
    unsigned short* __restrict__ Qb,
    unsigned short* __restrict__ Kb,
    unsigned short* __restrict__ Vt,
    int M, int N, int K, int mode)
{
    __shared__ __align__(16) unsigned short As[2 * TILE_ELEMS];
    __shared__ __align__(16) unsigned short Bs[2 * TILE_ELEMS];

    const int tid  = threadIdx.x;
    const int lane = tid & 31;
    const int wid  = tid >> 5;
    const int half = lane >> 4;
    const int r    = lane & 15;

    const int m0 = blockIdx.y * 128;
    const int n0 = blockIdx.x * 128;
    const int wm = (wid & 3) * 32;   // wave M offset within tile
    const int wn = (wid >> 2) * 64;  // wave N offset within tile

    v8f acc[2][4];
#pragma unroll
    for (int i = 0; i < 2; ++i)
#pragma unroll
        for (int j = 0; j < 4; ++j)
            acc[i][j] = (v8f){0.f,0.f,0.f,0.f,0.f,0.f,0.f,0.f};

    const int lrow = tid >> 1;         // 0..127
    const int lcol = (tid & 1) * 16;   // 0 or 16

    // per-thread staging addresses (buffer 0); buffer 1 is + TILE_BYTES
    const unsigned ldsA0 = lds_offset_of(As + lrow * LDT + lcol);
    const unsigned ldsB0 = lds_offset_of(Bs + lrow * LDT + lcol);
    const unsigned short* gA = A  + (size_t)(m0 + lrow) * K + lcol;
    const unsigned short* gB = Bm + (size_t)(n0 + lrow) * K + lcol;

    const int nsteps = K >> 5;

    // prologue: async-stage tile 0 into buffer 0
    async_ld32B(ldsA0, gA);
    async_ld32B(ldsB0, gB);

    for (int kk = 0; kk < nsteps; ++kk) {
        const int cur = kk & 1;

        wait_async0();         // this wave's tile-kk loads are in LDS
        __syncthreads();       // everyone's tile-kk loads are visible

        if (kk + 1 < nsteps) { // stage tile kk+1 into the other buffer
            const unsigned boff = (unsigned)((cur ^ 1) * TILE_BYTES);
            async_ld32B(ldsA0 + boff, gA + (size_t)(kk + 1) * 32);
            async_ld32B(ldsB0 + boff, gB + (size_t)(kk + 1) * 32);
        }

        const unsigned short* Ab = As + cur * TILE_ELEMS;
        const unsigned short* Bb = Bs + cur * TILE_ELEMS;

        Frag af[2], bf_[4];
#pragma unroll
        for (int i = 0; i < 2; ++i) {
            const unsigned short* p = Ab + (wm + i * 16 + r) * LDT + half * 8;
            af[i].u[0] = *(const uint4v*)(p);
            af[i].u[1] = *(const uint4v*)(p + 16);
        }
#pragma unroll
        for (int j = 0; j < 4; ++j) {
            const unsigned short* p = Bb + (wn + j * 16 + r) * LDT + half * 8;
            bf_[j].u[0] = *(const uint4v*)(p);
            bf_[j].u[1] = *(const uint4v*)(p + 16);
        }
#pragma unroll
        for (int i = 0; i < 2; ++i)
#pragma unroll
            for (int j = 0; j < 4; ++j)
                acc[i][j] = wmma_bf16(af[i], bf_[j], acc[i][j]);
        // no trailing barrier: next iteration's loads target the buffer we
        // just computed from, and are only issued after the next top barrier.
    }

    // ---- store ----
    if (mode == 0) {
#pragma unroll
        for (int i = 0; i < 2; ++i) {
            int row_base = m0 + wm + i * 16 + 8 * half;
#pragma unroll
            for (int j = 0; j < 4; ++j) {
                int col = n0 + wn + j * 16 + r;
                float bv = bias[col];
#pragma unroll
                for (int v = 0; v < 8; ++v)
                    Cout[(size_t)(row_base + v) * N + col] = acc[i][j][v] + bv;
            }
        }
    } else {
#pragma unroll
        for (int i = 0; i < 2; ++i) {
            int row_base = m0 + wm + i * 16 + 8 * half;
#pragma unroll
            for (int j = 0; j < 4; ++j) {
                int col = n0 + wn + j * 16 + r;     // 0..3071
                int sec = col >> 10;                // 0=Q 1=K 2=V
                int e   = col & 1023;
                int h   = e >> 6;
                int d   = e & 63;
                float bv = bias[col];
#pragma unroll
                for (int v = 0; v < 8; ++v) {
                    int m = row_base + v;           // 0..8191
                    int b = m >> 10;
                    int s = m & 1023;
                    unsigned short val = f2bf(acc[i][j][v] + bv);
                    size_t bh = (size_t)(b * HH + h);
                    if (sec == 0)      Qb[(bh * SS + s) * DD + d] = val;
                    else if (sec == 1) Kb[(bh * SS + s) * DD + d] = val;
                    else               Vt[(bh * DD + d) * SS + s] = val;
                }
            }
        }
    }
}

// ---------------------------------------------------------------------------
// Attention: per block = (b, h, 16-query tile)
//  scores = Q K^T / 8 -> LDS f32 ; softmax (writes f32 attn_weights out)
//  -> bf16 P in LDS ; out = P * V via Vt (NT), ds_add_f32 K-split reduction
// ---------------------------------------------------------------------------
__global__ __launch_bounds__(256) void attn_kernel(
    const unsigned short* __restrict__ Qb,
    const unsigned short* __restrict__ Kb,
    const unsigned short* __restrict__ Vt,
    float* __restrict__ attnw,
    unsigned short* __restrict__ AO)
{
    __shared__ __align__(16) float          sS[16 * 1024];    // 64 KB scores
    __shared__ __align__(16) unsigned short sP[16 * 1024];    // 32 KB probs (bf16)
    __shared__ __align__(16) float          sOut[16 * 64];    // 4 KB

    const int tid  = threadIdx.x;
    const int lane = tid & 31;
    const int wid  = tid >> 5;
    const int half = lane >> 4;
    const int r    = lane & 15;

    const int q0 = blockIdx.x * 16;
    const int h  = blockIdx.y;
    const int b  = blockIdx.z;
    const size_t bh = (size_t)(b * HH + h);

    const unsigned short* Qbase = Qb + bh * SS * DD;
    const unsigned short* Kbase = Kb + bh * SS * DD;
    const unsigned short* Vbase = Vt + bh * DD * SS;

    for (int i = tid; i < 16 * 64; i += 256) sOut[i] = 0.f;

    // Q fragments for the two 32-wide K chunks of D=64
    Frag qf[2];
    {
        const unsigned short* qrow = Qbase + (size_t)(q0 + r) * DD;
#pragma unroll
        for (int kk = 0; kk < 2; ++kk) {
            qf[kk].u[0] = *(const uint4v*)(qrow + kk * 32 + half * 8);
            qf[kk].u[1] = *(const uint4v*)(qrow + kk * 32 + 16 + half * 8);
        }
    }

    // ---- scores: 64 key tiles, 8 per wave ----
#pragma unroll 1
    for (int t = 0; t < 8; ++t) {
        int nt = wid * 8 + t;
        const unsigned short* krow = Kbase + (size_t)(nt * 16 + r) * DD;
        if (t + 1 < 8)  // pull next K-row group toward the WGP (global_prefetch_b8)
            __builtin_prefetch(krow + 16 * DD, 0, 0);
        Frag kf0, kf1;
        kf0.u[0] = *(const uint4v*)(krow + half * 8);
        kf0.u[1] = *(const uint4v*)(krow + 16 + half * 8);
        kf1.u[0] = *(const uint4v*)(krow + 32 + half * 8);
        kf1.u[1] = *(const uint4v*)(krow + 48 + half * 8);
        v8f a = (v8f){0.f,0.f,0.f,0.f,0.f,0.f,0.f,0.f};
        a = wmma_bf16(qf[0], kf0, a);
        a = wmma_bf16(qf[1], kf1, a);
        int col = nt * 16 + r;
#pragma unroll
        for (int v = 0; v < 8; ++v)
            sS[(8 * half + v) * 1024 + col] = a[v] * 0.125f; // 1/sqrt(64)
    }
    __syncthreads();

    // ---- softmax: 16 threads per query row, 64 cols each ----
    {
        const int row = tid >> 4;
        const int seg = tid & 15;
        float* srow = sS + row * 1024 + seg * 64;

        float mx = -3.402823466e+38f;
#pragma unroll 4
        for (int c = 0; c < 64; ++c) mx = fmaxf(mx, srow[c]);
#pragma unroll
        for (int msk = 1; msk < 16; msk <<= 1) mx = fmaxf(mx, __shfl_xor(mx, msk, 32));

        float sum = 0.f;
#pragma unroll 4
        for (int c = 0; c < 64; ++c) { float e = __expf(srow[c] - mx); srow[c] = e; sum += e; }
#pragma unroll
        for (int msk = 1; msk < 16; msk <<= 1) sum += __shfl_xor(sum, msk, 32);
        float inv = 1.f / sum;

        float* wout = attnw + (bh * SS + (q0 + row)) * (size_t)SS + seg * 64;
        unsigned short* prow = sP + row * 1024 + seg * 64;
#pragma unroll 2
        for (int c = 0; c < 64; c += 4) {
            float p0 = srow[c + 0] * inv;
            float p1 = srow[c + 1] * inv;
            float p2 = srow[c + 2] * inv;
            float p3 = srow[c + 3] * inv;
            float4 o; o.x = p0; o.y = p1; o.z = p2; o.w = p3;
            *(float4*)(wout + c) = o;                 // f32 attn_weights output
            uint2 pk; pk.x = pack2bf(p0, p1); pk.y = pack2bf(p2, p3);
            *(uint2*)(prow + c) = pk;                 // bf16 P for P*V
        }
    }
    __syncthreads();

    // ---- P * V : wave -> (d-tile = wid&3, K-half = wid>>2), ds_add reduce ----
    {
        const int dnt = wid & 3;
        const int kh  = wid >> 2;
        const unsigned short* vrow = Vbase + (size_t)(dnt * 16 + r) * SS;
        v8f a = (v8f){0.f,0.f,0.f,0.f,0.f,0.f,0.f,0.f};
#pragma unroll 1
        for (int kk = kh * 16; kk < kh * 16 + 16; ++kk) {
            Frag pf, vf;
            const unsigned short* prow = sP + r * 1024 + kk * 32;
            pf.u[0] = *(const uint4v*)(prow + half * 8);
            pf.u[1] = *(const uint4v*)(prow + 16 + half * 8);
            vf.u[0] = *(const uint4v*)(vrow + kk * 32 + half * 8);
            vf.u[1] = *(const uint4v*)(vrow + kk * 32 + 16 + half * 8);
            a = wmma_bf16(pf, vf, a);
        }
#pragma unroll
        for (int v = 0; v < 8; ++v)
            atomicAdd(&sOut[(8 * half + v) * 64 + dnt * 16 + r], a[v]);
    }
    __syncthreads();

    // ---- write attn output tile as bf16 into AO [B,S,E] (e = h*64+d) ----
    for (int i = tid; i < 16 * 64; i += 256) {
        int qrow = i >> 6;
        int d    = i & 63;
        AO[((size_t)b * SS + q0 + qrow) * EE + h * DD + d] = f2bf(sOut[i]);
    }
}

// ---------------------------------------------------------------------------
// Launch
// ---------------------------------------------------------------------------
extern "C" void kernel_launch(void* const* d_in, const int* in_sizes, int n_in,
                              void* d_out, int out_size, void* d_ws, size_t ws_size,
                              hipStream_t stream) {
    (void)in_sizes; (void)n_in; (void)out_size; (void)ws_size;

    const float* q    = (const float*)d_in[0];
    const float* Wqkv = (const float*)d_in[3];
    const float* bqkv = (const float*)d_in[4];
    const float* Wout = (const float*)d_in[5];
    const float* bout = (const float*)d_in[6];

    float* out   = (float*)d_out;                       // [B,S,E]
    float* attnw = out + (size_t)BB * SS * EE;          // [B,H,S,S]

    char* ws = (char*)d_ws;
    unsigned short* Xb    = (unsigned short*)(ws + XB_OFF);
    unsigned short* Wqkvb = (unsigned short*)(ws + WQKV_OFF);
    unsigned short* Woutb = (unsigned short*)(ws + WOUT_OFF);
    unsigned short* Qb    = (unsigned short*)(ws + QB_OFF);
    unsigned short* Kb    = (unsigned short*)(ws + KB_OFF);
    unsigned short* Vt    = (unsigned short*)(ws + VT_OFF);
    unsigned short* AO    = (unsigned short*)(ws + AO_OFF);

    // 1) bf16 conversions
    cvt_f32_bf16<<<(MM * EE / 4) / 256, 256, 0, stream>>>(q, Xb, MM * EE / 4);
    cvt_f32_bf16<<<(3 * EE * EE / 4) / 256, 256, 0, stream>>>(Wqkv, Wqkvb, 3 * EE * EE / 4);
    cvt_f32_bf16<<<(EE * EE / 4) / 256, 256, 0, stream>>>(Wout, Woutb, EE * EE / 4);

    // 2) QKV projection: [8192,1024] x [3072,1024]^T, scatter to Qb/Kb/Vt
    gemm_bf16_nt<<<dim3(3 * EE / 128, MM / 128), 256, 0, stream>>>(
        Xb, Wqkvb, bqkv, nullptr, Qb, Kb, Vt, MM, 3 * EE, EE, 1);

    // 3) attention (scores, softmax -> attn_weights, P*V -> AO)
    attn_kernel<<<dim3(SS / 16, HH, BB), 256, 0, stream>>>(Qb, Kb, Vt, attnw, AO);

    // 4) output projection: [8192,1024] x [1024,1024]^T + bias -> out
    gemm_bf16_nt<<<dim3(EE / 128, MM / 128), 256, 0, stream>>>(
        AO, Woutb, bout, out, nullptr, nullptr, nullptr, MM, EE, EE, 0);
}
